// Decoder_50294067036562
// MI455X (gfx1250) — compile-verified, matrix operand order checked by reading, not run on hardware
//
#include <hip/hip_runtime.h>
#include <hip/hip_bf16.h>
#include <math.h>

// MI455X / gfx1250, wave32. bf16 WMMA GEMMs (async-LDS double-buffered) + fp32 recurrence.
// B=512, T-1=63, DH=EH=512.

typedef __attribute__((ext_vector_type(16))) __bf16 v16bf;
typedef __attribute__((ext_vector_type(8)))  float  v8f;
typedef __attribute__((ext_vector_type(8)))  __bf16 bf16x8;
typedef __attribute__((ext_vector_type(4)))  int    v4i;

#define BN_   512
#define TM1   63
#define DH_   512
#define EH_   512
#define NG_   2048   // 4*DH

#if defined(__has_builtin)
#if __has_builtin(__builtin_amdgcn_global_load_async_to_lds_b128)
#define HAVE_ASYNC_LDS 1
#endif
#endif
#ifndef HAVE_ASYNC_LDS
#define HAVE_ASYNC_LDS 0
#endif

__device__ __forceinline__ float sigmoidf_(float x) { return 1.0f / (1.0f + __expf(-x)); }

// one 16-byte global->LDS move per lane via the async (ASYNCcnt-tracked) pipe
__device__ __forceinline__ void copy16_g2l(const __bf16* g, __bf16* l) {
#if HAVE_ASYNC_LDS
  __builtin_amdgcn_global_load_async_to_lds_b128(
      (__attribute__((address_space(1))) v4i*)(const void*)g,
      (__attribute__((address_space(3))) v4i*)(void*)l, 0, 0);
#else
  *(bf16x8*)l = *(const bf16x8*)g;
#endif
}

template <int N>
__device__ __forceinline__ void wait_async() {
#if HAVE_ASYNC_LDS
#if __has_builtin(__builtin_amdgcn_s_wait_asynccnt)
  __builtin_amdgcn_s_wait_asynccnt(N);
#else
  asm volatile("s_wait_asynccnt %0" ::"i"(N) : "memory");
#endif
#endif
}

// ---------------- conversion / prep ----------------
__global__ void k_f32_to_bf16(const float* __restrict__ s, __bf16* __restrict__ d, int n) {
  int i = blockIdx.x * blockDim.x + threadIdx.x;
  if (i < n) d[i] = (__bf16)s[i];
}

// attn_W1 [EH, 2DH+EH] -> W1dc [EH, 2DH] bf16, W1enc [EH, EH] bf16
__global__ void k_prep_w1(const float* __restrict__ W1, __bf16* __restrict__ Wdc,
                          __bf16* __restrict__ Wenc) {
  int i = blockIdx.x * blockDim.x + threadIdx.x;
  if (i >= EH_ * (2 * DH_ + EH_)) return;
  int f = i / (2 * DH_ + EH_), e = i % (2 * DH_ + EH_);
  __bf16 v = (__bf16)W1[i];
  if (e < 2 * DH_) Wdc[f * (2 * DH_) + e] = v;
  else             Wenc[f * EH_ + (e - 2 * DH_)] = v;
}

__global__ void k_prep_bias(const float* __restrict__ a, const float* __restrict__ b,
                            float* __restrict__ o, int n) {
  int i = blockIdx.x * blockDim.x + threadIdx.x;
  if (i < n) o[i] = a[i] + b[i];
}

// ---------------- bf16 WMMA GEMM: C[M,N] = A[M,K] @ W[N,K]^T (+bias[N]) ----------------
// Block tile 128x64 (8 waves, 4x2 grid of 32x32 wave tiles), BK=32,
// double-buffered LDS fed by async global->LDS b128 copies; last tile peeled
// so the steady-state loop body is branch-free.
#define GBM 128
#define GBN 64
#define GBK 32
#define LSTR 48   // LDS row stride in bf16 (96B -> keeps 16B alignment)

__global__ __launch_bounds__(256) void k_gemm_bf16(
    const __bf16* __restrict__ A, const __bf16* __restrict__ W,
    const float* __restrict__ bias, float* __restrict__ C,
    int M, int N, int K)
{
  __shared__ __bf16 As[2][GBM * LSTR];
  __shared__ __bf16 Bs[2][GBN * LSTR];

  const int tid  = threadIdx.x;
  const int lane = tid & 31;
  const int wave = tid >> 5;
  const int wm   = (wave & 3) * 32;   // wave M offset in block tile
  const int wn   = (wave >> 2) * 32;  // wave N offset in block tile
  const size_t bm = (size_t)blockIdx.y * GBM;
  const size_t bn = (size_t)blockIdx.x * GBN;

  // copy mapping: each thread moves 3x16B per K-tile (2 A rows + 1 B row)
  const int car = tid >> 2;          // 0..63
  const int cac = (tid & 3) * 8;     // 0,8,16,24 (bf16)

  const __bf16* gA0 = A + (bm + car) * (size_t)K + cac;
  const __bf16* gA1 = A + (bm + car + 64) * (size_t)K + cac;
  const __bf16* gB  = W + (bn + car) * (size_t)K + cac;
  const int lA0 = car * LSTR + cac;
  const int lA1 = (car + 64) * LSTR + cac;
  const int lB  = car * LSTR + cac;

  v8f acc00 = {}, acc01 = {}, acc10 = {}, acc11 = {};

  const int lhalf = (lane < 16) ? 0 : 8;   // documented 16-bit fragment K striping
  const int lm    = lane & 15;
  const int a0row = (wm + lm) * LSTR;
  const int a1row = (wm + 16 + lm) * LSTR;
  const int b0row = (wn + lm) * LSTR;
  const int b1row = (wn + 16 + lm) * LSTR;

  auto compute_tile = [&](int cur) {
    union { v16bf v; bf16x8 h[2]; } fa0, fa1, fb0, fb1;
    const __bf16* Ab = As[cur];
    const __bf16* Bb = Bs[cur];
    fa0.h[0] = *(const bf16x8*)&Ab[a0row + lhalf];
    fa0.h[1] = *(const bf16x8*)&Ab[a0row + 16 + lhalf];
    fa1.h[0] = *(const bf16x8*)&Ab[a1row + lhalf];
    fa1.h[1] = *(const bf16x8*)&Ab[a1row + 16 + lhalf];
    fb0.h[0] = *(const bf16x8*)&Bb[b0row + lhalf];
    fb0.h[1] = *(const bf16x8*)&Bb[b0row + 16 + lhalf];
    fb1.h[0] = *(const bf16x8*)&Bb[b1row + lhalf];
    fb1.h[1] = *(const bf16x8*)&Bb[b1row + 16 + lhalf];
    acc00 = __builtin_amdgcn_wmma_f32_16x16x32_bf16(false, fa0.v, false, fb0.v, (short)0, acc00, false, false);
    acc01 = __builtin_amdgcn_wmma_f32_16x16x32_bf16(false, fa0.v, false, fb1.v, (short)0, acc01, false, false);
    acc10 = __builtin_amdgcn_wmma_f32_16x16x32_bf16(false, fa1.v, false, fb0.v, (short)0, acc10, false, false);
    acc11 = __builtin_amdgcn_wmma_f32_16x16x32_bf16(false, fa1.v, false, fb1.v, (short)0, acc11, false, false);
  };

  // prefetch tile 0
  copy16_g2l(gA0, &As[0][lA0]);
  copy16_g2l(gA1, &As[0][lA1]);
  copy16_g2l(gB,  &Bs[0][lB]);

  const int ntiles = K / GBK;
  // steady state: always prefetch next, wait for current (branch-free body)
  for (int i = 0; i < ntiles - 1; ++i) {
    const int cur  = i & 1;
    const int koff = (i + 1) * GBK;
    copy16_g2l(gA0 + koff, &As[cur ^ 1][lA0]);
    copy16_g2l(gA1 + koff, &As[cur ^ 1][lA1]);
    copy16_g2l(gB  + koff, &Bs[cur ^ 1][lB]);
    wait_async<3>();   // current tile's 3 copies done (in-order), next in flight
    __syncthreads();
    compute_tile(cur);
    __syncthreads();
  }
  // final tile
  wait_async<0>();
  __syncthreads();
  compute_tile((ntiles - 1) & 1);

  const int erow = (lane < 16) ? 0 : 8;
  const int ecol = lane & 15;
#define STORE_TILE(ACC, MOFF, NOFF)                                              \
  {                                                                              \
    const size_t col = bn + wn + (NOFF) + ecol;                                  \
    const float bv = bias ? bias[col] : 0.0f;                                    \
    _Pragma("unroll")                                                            \
    for (int r = 0; r < 8; ++r)                                                  \
      C[(bm + wm + (MOFF) + erow + r) * (size_t)N + col] = (ACC)[r] + bv;        \
  }
  STORE_TILE(acc00, 0, 0)
  STORE_TILE(acc01, 0, 16)
  STORE_TILE(acc10, 16, 0)
  STORE_TILE(acc11, 16, 16)
#undef STORE_TILE
}

// ---------------- fused attention: scores -> softmax -> ctx -> y_tilde ----------------
__global__ __launch_bounds__(256) void k_attn(
    const float* __restrict__ P, const float* __restrict__ q,
    const float* __restrict__ Xenc, const float* __restrict__ w2,
    const float* __restrict__ b2, const float* __restrict__ fcW,
    const float* __restrict__ fcb, const float* __restrict__ y_prev,
    float* __restrict__ ctx, float* __restrict__ y_tilde, int tstep)
{
  __shared__ __align__(16) float qs[EH_];
  __shared__ __align__(16) float w2s[EH_];
  __shared__ float sc[TM1 + 1];
  __shared__ float bet[TM1 + 1];
  __shared__ float red[256];

  const int b    = blockIdx.x;
  const int tid  = threadIdx.x;
  const int lane = tid & 31;
  const int wave = tid >> 5;

  qs[tid]        = q[b * EH_ + tid];
  qs[tid + 256]  = q[b * EH_ + tid + 256];
  w2s[tid]       = w2[tid];
  w2s[tid + 256] = w2[tid + 256];
  __syncthreads();

  // scores[t] = w2 . tanh(P[b,t,:] + q[b,:]) + b2  — one wave per t, float4 streams
  const float4* qs4 = (const float4*)qs;
  const float4* w24 = (const float4*)w2s;
  for (int t = wave; t < TM1; t += 8) {
    const float4* Pr = (const float4*)(P + ((size_t)b * TM1 + t) * EH_);
    float a = 0.f;
    for (int e = lane; e < EH_ / 4; e += 32) {
      float4 p = Pr[e], qq = qs4[e], ww = w24[e];
      a += ww.x * tanhf(p.x + qq.x) + ww.y * tanhf(p.y + qq.y) +
           ww.z * tanhf(p.z + qq.z) + ww.w * tanhf(p.w + qq.w);
    }
    for (int o = 16; o; o >>= 1) a += __shfl_xor(a, o, 32);
    if (lane == 0) sc[t] = a + b2[0];
  }
  __syncthreads();

  // softmax over 63 scores (wave 0)
  if (wave == 0) {
    float s0 = (lane < TM1) ? sc[lane] : -INFINITY;
    float s1 = (lane + 32 < TM1) ? sc[lane + 32] : -INFINITY;
    float m = fmaxf(s0, s1);
    for (int o = 16; o; o >>= 1) m = fmaxf(m, __shfl_xor(m, o, 32));
    float e0 = (lane < TM1) ? __expf(s0 - m) : 0.f;
    float e1 = (lane + 32 < TM1) ? __expf(s1 - m) : 0.f;
    float s = e0 + e1;
    for (int o = 16; o; o >>= 1) s += __shfl_xor(s, o, 32);
    float inv = 1.f / s;
    if (lane < TM1)      bet[lane]      = e0 * inv;
    if (lane + 32 < TM1) bet[lane + 32] = e1 * inv;
  }
  __syncthreads();

  // ctx[b,e] = sum_t beta[t]*X[b,t,e]; partial for y_tilde = fcW[:EH].ctx
  float part = 0.f;
  for (int e = tid; e < EH_; e += 256) {
    const float* Xb = Xenc + (size_t)b * TM1 * EH_ + e;
    float a = 0.f;
#pragma unroll 9
    for (int t = 0; t < TM1; ++t) a += bet[t] * Xb[(size_t)t * EH_];
    ctx[b * EH_ + e] = a;
    part += a * fcW[e];
  }
  red[tid] = part;
  __syncthreads();
  for (int s = 128; s; s >>= 1) { if (tid < s) red[tid] += red[tid + s]; __syncthreads(); }
  if (tid == 0)
    y_tilde[b] = red[0] + fcW[EH_] * y_prev[b * TM1 + tstep] + fcb[0];
}

// ---------------- LSTM pointwise update (emits bf16 state for next-step GEMMs) ----------------
__global__ __launch_bounds__(256) void k_lstm(
    const float* __restrict__ gates, const float* __restrict__ Wih,
    const float* __restrict__ y_tilde,
    float* __restrict__ c, float* __restrict__ d,
    __bf16* __restrict__ d_bf, __bf16* __restrict__ dc_bf)
{
  int idx = blockIdx.x * blockDim.x + threadIdx.x;
  if (idx >= BN_ * DH_) return;
  int b = idx >> 9, k = idx & (DH_ - 1);
  float yt = y_tilde[b];
  const float* g = gates + (size_t)b * NG_;   // (b_ih+b_hh) folded by GEMM epilogue
  float gi = g[k]           + yt * Wih[k];
  float gf = g[DH_ + k]     + yt * Wih[DH_ + k];
  float gg = g[2 * DH_ + k] + yt * Wih[2 * DH_ + k];
  float go = g[3 * DH_ + k] + yt * Wih[3 * DH_ + k];
  float cn = sigmoidf_(gf) * c[idx] + sigmoidf_(gi) * tanhf(gg);
  float dn = sigmoidf_(go) * tanhf(cn);
  c[idx] = cn;
  d[idx] = dn;
  __bf16 db = (__bf16)dn, cb = (__bf16)cn;
  d_bf[idx] = db;
  dc_bf[(size_t)b * (2 * DH_) + k]       = db;
  dc_bf[(size_t)b * (2 * DH_) + DH_ + k] = cb;
}

// ---------------- final heads: out order (beta, sigma, gamma) ----------------
__global__ __launch_bounds__(256) void k_heads(
    const float* __restrict__ d, const float* __restrict__ ctx,
    const float* __restrict__ bW, const float* __restrict__ bb,
    const float* __restrict__ gW, const float* __restrict__ gb,
    const float* __restrict__ sW, const float* __restrict__ sb,
    float* __restrict__ out)
{
  __shared__ float r0[256], r1[256], r2[256];
  int b = blockIdx.x, tid = threadIdx.x;
  float a0 = 0, a1 = 0, a2 = 0;
  for (int j = tid; j < DH_ + EH_; j += 256) {
    float v = (j < DH_) ? d[b * DH_ + j] : ctx[b * EH_ + (j - DH_)];
    a0 += v * bW[j];  // beta
    a1 += v * sW[j];  // sigma
    a2 += v * gW[j];  // gamma
  }
  r0[tid] = a0; r1[tid] = a1; r2[tid] = a2;
  __syncthreads();
  for (int s = 128; s; s >>= 1) {
    if (tid < s) { r0[tid] += r0[tid + s]; r1[tid] += r1[tid + s]; r2[tid] += r2[tid + s]; }
    __syncthreads();
  }
  if (tid == 0) {
    out[b]           = r0[0] + bb[0];
    out[BN_ + b]     = r1[0] + sb[0];
    out[2 * BN_ + b] = r2[0] + gb[0];
  }
}

// ---------------- host launcher ----------------
extern "C" void kernel_launch(void* const* d_in, const int* in_sizes, int n_in,
                              void* d_out, int out_size, void* d_ws, size_t ws_size,
                              hipStream_t stream) {
  const float* X   = (const float*)d_in[0];   // [512,63,512]
  const float* yp  = (const float*)d_in[1];   // [512,63]
  const float* W1  = (const float*)d_in[2];   // [512,1536]
  const float* b1  = (const float*)d_in[3];   // [512]
  const float* W2  = (const float*)d_in[4];   // [1,512]
  const float* b2  = (const float*)d_in[5];   // [1]
  const float* fcW = (const float*)d_in[6];   // [1,513]
  const float* fcb = (const float*)d_in[7];   // [1]
  const float* Wih = (const float*)d_in[8];   // [2048,1]
  const float* Whh = (const float*)d_in[9];   // [2048,512]
  const float* bih = (const float*)d_in[10];  // [2048]
  const float* bhh = (const float*)d_in[11];  // [2048]
  const float* bW  = (const float*)d_in[12];
  const float* bb  = (const float*)d_in[13];
  const float* gW  = (const float*)d_in[14];
  const float* gb  = (const float*)d_in[15];
  const float* sW  = (const float*)d_in[16];
  const float* sb  = (const float*)d_in[17];
  float* out = (float*)d_out;

  char* ws = (char*)d_ws;
  auto alloc = [&](size_t bytes) -> char* {
    char* p = ws;
    ws += (bytes + 255) & ~(size_t)255;
    return p;
  };
  __bf16* Xbf   = (__bf16*)alloc((size_t)BN_ * TM1 * EH_ * 2);
  __bf16* W1enc = (__bf16*)alloc((size_t)EH_ * EH_ * 2);
  __bf16* W1dc  = (__bf16*)alloc((size_t)EH_ * 2 * DH_ * 2);
  __bf16* Whhbf = (__bf16*)alloc((size_t)NG_ * DH_ * 2);
  float*  biasg = (float*)alloc((size_t)NG_ * 4);
  float*  P     = (float*)alloc((size_t)BN_ * TM1 * EH_ * 4);
  float*  q     = (float*)alloc((size_t)BN_ * EH_ * 4);
  float*  gates = (float*)alloc((size_t)BN_ * NG_ * 4);
  float*  dst   = (float*)alloc((size_t)BN_ * DH_ * 4);
  float*  cst   = (float*)alloc((size_t)BN_ * DH_ * 4);
  __bf16* dbf   = (__bf16*)alloc((size_t)BN_ * DH_ * 2);
  __bf16* dcbf  = (__bf16*)alloc((size_t)BN_ * 2 * DH_ * 2);
  float*  ctx   = (float*)alloc((size_t)BN_ * EH_ * 4);
  float*  yt    = (float*)alloc((size_t)BN_ * 4);

  // zero-init recurrent state (graph-capture-safe)
  (void)hipMemsetAsync(dst,  0, (size_t)BN_ * DH_ * 4, stream);
  (void)hipMemsetAsync(cst,  0, (size_t)BN_ * DH_ * 4, stream);
  (void)hipMemsetAsync(dbf,  0, (size_t)BN_ * DH_ * 2, stream);
  (void)hipMemsetAsync(dcbf, 0, (size_t)BN_ * 2 * DH_ * 2, stream);

  // conversions
  int nX = BN_ * TM1 * EH_;
  k_f32_to_bf16<<<(nX + 255) / 256, 256, 0, stream>>>(X, Xbf, nX);
  k_f32_to_bf16<<<(NG_ * DH_ + 255) / 256, 256, 0, stream>>>(Whh, Whhbf, NG_ * DH_);
  k_prep_w1<<<(EH_ * (2 * DH_ + EH_) + 255) / 256, 256, 0, stream>>>(W1, W1dc, W1enc);
  k_prep_bias<<<(NG_ + 255) / 256, 256, 0, stream>>>(bih, bhh, biasg, NG_);

  // one-time: P = X_enc @ W1_enc^T + b1   [32256, 512]  (32256 = 252*128)
  k_gemm_bf16<<<dim3(EH_ / GBN, (BN_ * TM1) / GBM), 256, 0, stream>>>(
      Xbf, W1enc, b1, P, BN_ * TM1, EH_, EH_);

  // 63-step recurrence
  for (int t = 0; t < TM1; ++t) {
    // q = [d,c] @ W1_dc^T   [512,512], K=1024
    k_gemm_bf16<<<dim3(EH_ / GBN, BN_ / GBM), 256, 0, stream>>>(
        dcbf, W1dc, nullptr, q, BN_, EH_, 2 * DH_);
    // gates_pre = d @ W_hh^T + (b_ih+b_hh)   [512,2048], K=512
    k_gemm_bf16<<<dim3(NG_ / GBN, BN_ / GBM), 256, 0, stream>>>(
        dbf, Whhbf, biasg, gates, BN_, NG_, DH_);
    // attention + softmax + context + y_tilde
    k_attn<<<BN_, 256, 0, stream>>>(P, q, X, W2, b2, fcW, fcb, yp, ctx, yt, t);
    // LSTM pointwise update
    k_lstm<<<(BN_ * DH_ + 255) / 256, 256, 0, stream>>>(gates, Wih, yt, cst, dst, dbf, dcbf);
  }

  // heads: (y_beta, y_sigma, y_gamma)
  k_heads<<<BN_, 256, 0, stream>>>(dst, ctx, bW, bb, gW, gb, sW, sb, out);
}
